// GGNNRel_5669356831747
// MI455X (gfx1250) — compile-verified, matrix operand order checked by reading, not run on hardware
//
#include <hip/hip_runtime.h>
#include <hip/hip_bf16.h>

#define D_      512
#define NNODE   53
#define C_      51
#define R_      2048
#define RN_     (R_ * NNODE)       // 108544 = 848 * 128
#define OBJ_    151
#define ND_     (NNODE * D_)       // 27136 = 424 * 64
#define KC      64                 // K-chunk staged in LDS
#define LDSW    72                 // padded LDS row stride (halves) to spread banks

typedef _Float16 half8  __attribute__((ext_vector_type(8)));
typedef _Float16 half16 __attribute__((ext_vector_type(16)));
typedef float    float8 __attribute__((ext_vector_type(8)));

// ---- WMMA fragment helpers (CDNA5 wave32 16x16x32 f16 layout) ----
static __device__ __forceinline__ half16
frag_load(const _Float16* __restrict__ base, long row, int ld, int kb, int lh) {
    const _Float16* p = base + row * (long)ld + kb + lh * 8;
    half8 lo = *(const half8*)(p);
    half8 hi = *(const half8*)(p + 16);
    return __builtin_shufflevector(lo, hi, 0,1,2,3,4,5,6,7,8,9,10,11,12,13,14,15);
}

static __device__ __forceinline__ half16
frag_lds(const _Float16* buf, int row, int k, int lh) {
    const _Float16* p = buf + row * LDSW + k + lh * 8;
    half8 lo = *(const half8*)(p);
    half8 hi = *(const half8*)(p + 16);
    return __builtin_shufflevector(lo, hi, 0,1,2,3,4,5,6,7,8,9,10,11,12,13,14,15);
}

// Stage a 64(n) x KC(k) f16 weight tile into LDS; 256 threads x 32B each.
static __device__ __forceinline__ void
stage_w(_Float16* buf, const _Float16* __restrict__ W, long ldw,
        int nBase, long kc, int Ncols, int tid) {
    const int n  = tid >> 2;          // 0..63
    const int kq = (tid & 3) << 4;    // 0,16,32,48
    half8 a = {}, b = {};
    const int gn = nBase + n;
    if (gn < Ncols) {
        const _Float16* p = W + (long)gn * ldw + kc + kq;
        a = *(const half8*)(p);
        b = *(const half8*)(p + 8);
    }
    _Float16* q = buf + n * LDSW + kq;
    *(half8*)(q)     = a;
    *(half8*)(q + 8) = b;
}

static __device__ __forceinline__ float8 wmma32(half16 a, half16 b, float8 c) {
    return __builtin_amdgcn_wmma_f32_16x16x32_f16(false, a, false, b, (short)0, c, false, false);
}

static __device__ __forceinline__ float sigmoid_(float x) {
    return 1.0f / (1.0f + __expf(-x));
}

// ---- prep kernels ----
extern "C" __global__ void k_gather(const int* __restrict__ so,
                                    const float* __restrict__ matrix,
                                    float* __restrict__ Bg) {
    int i = blockIdx.x * 256 + threadIdx.x;
    if (i >= R_ * C_) return;
    int r = i / C_, c = i % C_;
    long s = so[2 * r], o = so[2 * r + 1];
    Bg[i] = matrix[(s * OBJ_ + o) * C_ + c];
}

extern "C" __global__ void k_fold(const float* __restrict__ w, _Float16* __restrict__ o) {
    int i = blockIdx.x * 256 + threadIdx.x;
    if (i >= D_ * D_) return;
    int n = i / D_, k = i % D_;
    o[i] = (_Float16)(w[(long)n * (2 * D_) + k] + w[(long)n * (2 * D_) + D_ + k]);
}

extern "C" __global__ void k_cvt(const float* __restrict__ s, _Float16* __restrict__ d, long n) {
    long i = blockIdx.x * 256L + threadIdx.x;
    if (i < n) d[i] = (_Float16)s[i];
}

// ---- av construction ----
extern "C" __global__ void __launch_bounds__(256)
k_av(const _Float16* __restrict__ h16, const float* __restrict__ Bg,
     _Float16* __restrict__ av16) {
    int r = blockIdx.x;
    int d = blockIdx.y * 256 + threadIdx.x;
    __shared__ float Bs[C_];
    if (threadIdx.x < C_) Bs[threadIdx.x] = Bg[r * C_ + threadIdx.x];
    __syncthreads();
    const long base = (long)r * NNODE * D_;
    float s01 = (float)h16[base + d] + (float)h16[base + D_ + d];
    float top = 0.f;
#pragma unroll 3
    for (int c = 0; c < C_; ++c)
        top += Bs[c] * (float)h16[base + (long)(2 + c) * D_ + d];
    av16[base + d]       = (_Float16)top;
    av16[base + D_ + d]  = (_Float16)top;
    for (int c = 0; c < C_; ++c)
        av16[base + (long)(2 + c) * D_ + d] = (_Float16)(Bs[c] * s01);
}

// Wave layout inside a 128Mx64N block: wave w -> mSub = w&3 (32 rows), nSub = w>>2 (32 cols).
// Each wave: 2 M-tiles x 2 N-tiles; every LDS B-fragment feeds 2 WMMAs.

// ---- fused gate GEMMs: Z=av@W3s.T, Rr=av@W4s.T, U=h@W3u.T ----
extern "C" __global__ void __launch_bounds__(256)
k_gates(const _Float16* __restrict__ av, const _Float16* __restrict__ h16,
        const _Float16* __restrict__ W3s, const _Float16* __restrict__ W4s,
        const _Float16* __restrict__ W3u,
        const float* __restrict__ b3w, const float* __restrict__ b3u,
        const float* __restrict__ b4w,
        _Float16* __restrict__ zv16, _Float16* __restrict__ rh16) {
    __shared__ _Float16 s3[2][64 * LDSW];
    __shared__ _Float16 s4[2][64 * LDSW];
    __shared__ _Float16 su[2][64 * LDSW];
    const int tid = threadIdx.x;
    const int lane = tid & 31, wave = tid >> 5;
    const int lm = lane & 15, lh = lane >> 4;
    const int mSub = wave & 3, nSub = wave >> 2;
    const long mBase = (long)blockIdx.x * 128 + mSub * 32;
    const int nBase = blockIdx.y * 64 + nSub * 32;
    const int rlB = nSub * 32;
    float8 aZ[2][2] = {}; float8 aR[2][2] = {}; float8 aU[2][2] = {};
    const long am0 = mBase + lm, am1 = mBase + 16 + lm;

    stage_w(s3[0], W3s, D_, blockIdx.y * 64, 0, D_, tid);
    stage_w(s4[0], W4s, D_, blockIdx.y * 64, 0, D_, tid);
    stage_w(su[0], W3u, D_, blockIdx.y * 64, 0, D_, tid);
    int buf = 0;
    for (int kc = 0; kc < D_; kc += KC) {
        // A-fragments for the whole chunk: issue before the barrier
        half16 fa[2][2], fh[2][2];
        fa[0][0] = frag_load(av,  am0, D_, kc,      lh);
        fa[0][1] = frag_load(av,  am0, D_, kc + 32, lh);
        fa[1][0] = frag_load(av,  am1, D_, kc,      lh);
        fa[1][1] = frag_load(av,  am1, D_, kc + 32, lh);
        fh[0][0] = frag_load(h16, am0, D_, kc,      lh);
        fh[0][1] = frag_load(h16, am0, D_, kc + 32, lh);
        fh[1][0] = frag_load(h16, am1, D_, kc,      lh);
        fh[1][1] = frag_load(h16, am1, D_, kc + 32, lh);
        __syncthreads();
        if (kc + KC < D_) {   // stage next chunk while computing this one
            stage_w(s3[buf ^ 1], W3s, D_, blockIdx.y * 64, kc + KC, D_, tid);
            stage_w(s4[buf ^ 1], W4s, D_, blockIdx.y * 64, kc + KC, D_, tid);
            stage_w(su[buf ^ 1], W3u, D_, blockIdx.y * 64, kc + KC, D_, tid);
        }
#pragma unroll
        for (int g = 0; g < 4; ++g) {
            const int t = g >> 1, k2 = g & 1;
            const int rl = rlB + t * 16 + lm;
            half16 b3 = frag_lds(s3[buf], rl, k2 * 32, lh);
            half16 b4 = frag_lds(s4[buf], rl, k2 * 32, lh);
            half16 bu = frag_lds(su[buf], rl, k2 * 32, lh);
#pragma unroll
            for (int mi = 0; mi < 2; ++mi) {
                aZ[mi][t] = wmma32(fa[mi][k2], b3, aZ[mi][t]);
                aR[mi][t] = wmma32(fa[mi][k2], b4, aR[mi][t]);
                aU[mi][t] = wmma32(fh[mi][k2], bu, aU[mi][t]);
            }
        }
        buf ^= 1;
    }
#pragma unroll
    for (int t = 0; t < 2; ++t) {
        int n = nBase + t * 16 + lm;
        float bu = b3u[n], bz = b3w[n], br = b4w[n];
#pragma unroll
        for (int mi = 0; mi < 2; ++mi) {
#pragma unroll
            for (int v = 0; v < 8; ++v) {
                long m = mBase + mi * 16 + v + 8 * lh;
                long idx = m * D_ + n;
                float u = aU[mi][t][v] + bu;
                float z = sigmoid_(aZ[mi][t][v] + bz + u);
                float r = sigmoid_(aR[mi][t][v] + br + u);
                float hf = (float)h16[idx];
                zv16[idx] = (_Float16)z;
                rh16[idx] = (_Float16)(r * hf);
            }
        }
    }
}

// ---- fused update GEMMs: H=av@W5s.T, G=rh@W5u.T ; h16 updated in place ----
extern "C" __global__ void __launch_bounds__(256)
k_update(const _Float16* __restrict__ av, const _Float16* __restrict__ rh,
         const _Float16* __restrict__ W5s, const _Float16* __restrict__ W5u,
         const float* __restrict__ b5w, const float* __restrict__ b5u,
         const _Float16* __restrict__ zv16, _Float16* __restrict__ h16) {
    __shared__ _Float16 s5[2][64 * LDSW];
    __shared__ _Float16 sg[2][64 * LDSW];
    const int tid = threadIdx.x;
    const int lane = tid & 31, wave = tid >> 5;
    const int lm = lane & 15, lh = lane >> 4;
    const int mSub = wave & 3, nSub = wave >> 2;
    const long mBase = (long)blockIdx.x * 128 + mSub * 32;
    const int nBase = blockIdx.y * 64 + nSub * 32;
    const int rlB = nSub * 32;
    float8 aH[2][2] = {}; float8 aG[2][2] = {};
    const long am0 = mBase + lm, am1 = mBase + 16 + lm;

    stage_w(s5[0], W5s, D_, blockIdx.y * 64, 0, D_, tid);
    stage_w(sg[0], W5u, D_, blockIdx.y * 64, 0, D_, tid);
    int buf = 0;
    for (int kc = 0; kc < D_; kc += KC) {
        half16 fa[2][2], fr[2][2];
        fa[0][0] = frag_load(av, am0, D_, kc,      lh);
        fa[0][1] = frag_load(av, am0, D_, kc + 32, lh);
        fa[1][0] = frag_load(av, am1, D_, kc,      lh);
        fa[1][1] = frag_load(av, am1, D_, kc + 32, lh);
        fr[0][0] = frag_load(rh, am0, D_, kc,      lh);
        fr[0][1] = frag_load(rh, am0, D_, kc + 32, lh);
        fr[1][0] = frag_load(rh, am1, D_, kc,      lh);
        fr[1][1] = frag_load(rh, am1, D_, kc + 32, lh);
        __syncthreads();
        if (kc + KC < D_) {
            stage_w(s5[buf ^ 1], W5s, D_, blockIdx.y * 64, kc + KC, D_, tid);
            stage_w(sg[buf ^ 1], W5u, D_, blockIdx.y * 64, kc + KC, D_, tid);
        }
#pragma unroll
        for (int g = 0; g < 4; ++g) {
            const int t = g >> 1, k2 = g & 1;
            const int rl = rlB + t * 16 + lm;
            half16 b5 = frag_lds(s5[buf], rl, k2 * 32, lh);
            half16 bg = frag_lds(sg[buf], rl, k2 * 32, lh);
#pragma unroll
            for (int mi = 0; mi < 2; ++mi) {
                aH[mi][t] = wmma32(fa[mi][k2], b5, aH[mi][t]);
                aG[mi][t] = wmma32(fr[mi][k2], bg, aG[mi][t]);
            }
        }
        buf ^= 1;
    }
#pragma unroll
    for (int t = 0; t < 2; ++t) {
        int n = nBase + t * 16 + lm;
        float bh = b5w[n] + b5u[n];
#pragma unroll
        for (int mi = 0; mi < 2; ++mi) {
#pragma unroll
            for (int v = 0; v < 8; ++v) {
                long m = mBase + mi * 16 + v + 8 * lh;
                long idx = m * D_ + n;
                float hv = tanhf(aH[mi][t][v] + aG[mi][t][v] + bh);
                float z = (float)zv16[idx];
                float hold = (float)h16[idx];
                h16[idx] = (_Float16)((1.0f - z) * hold + z * hv);
            }
        }
    }
}

// ---- generic WMMA GEMM: out = act(A[M,K] @ W[N,K]^T + bias); A split at ksplit ----
extern "C" __global__ void __launch_bounds__(256)
k_gemm(const _Float16* __restrict__ A, const _Float16* __restrict__ A2, int ksplit, int lda,
       const _Float16* __restrict__ W, int ldw, const float* __restrict__ bias,
       _Float16* __restrict__ out16, float* __restrict__ out32, int ldc,
       int M, int Ncols, int K, int relu) {
    __shared__ _Float16 sw[2][64 * LDSW];
    const int tid = threadIdx.x;
    const int lane = tid & 31, wave = tid >> 5;
    const int lm = lane & 15, lh = lane >> 4;
    const int mSub = wave & 3, nSub = wave >> 2;
    const long mBase = (long)blockIdx.x * 128 + mSub * 32;
    const int nBase = blockIdx.y * 64 + nSub * 32;
    const int rlB = nSub * 32;
    float8 acc[2][2] = {};
    long am0 = mBase + lm;      if (am0 >= M) am0 = M - 1;
    long am1 = mBase + 16 + lm; if (am1 >= M) am1 = M - 1;

    stage_w(sw[0], W, ldw, blockIdx.y * 64, 0, Ncols, tid);
    int buf = 0;
    for (int kc = 0; kc < K; kc += KC) {
        const _Float16* src = A;
        long kk = kc;
        if (kc >= ksplit) { src = A2; kk = kc - ksplit; }
        half16 fa[2][2];
        fa[0][0] = frag_load(src, am0, lda, (int)kk,      lh);
        fa[0][1] = frag_load(src, am0, lda, (int)kk + 32, lh);
        fa[1][0] = frag_load(src, am1, lda, (int)kk,      lh);
        fa[1][1] = frag_load(src, am1, lda, (int)kk + 32, lh);
        __syncthreads();
        if (kc + KC < K)
            stage_w(sw[buf ^ 1], W, ldw, blockIdx.y * 64, kc + KC, Ncols, tid);
#pragma unroll
        for (int g = 0; g < 4; ++g) {
            const int t = g >> 1, k2 = g & 1;
            const int rl = rlB + t * 16 + lm;
            half16 cw = frag_lds(sw[buf], rl, k2 * 32, lh);
#pragma unroll
            for (int mi = 0; mi < 2; ++mi)
                acc[mi][t] = wmma32(fa[mi][k2], cw, acc[mi][t]);
        }
        buf ^= 1;
    }
#pragma unroll
    for (int t = 0; t < 2; ++t) {
        int n = nBase + t * 16 + lm;
        if (n >= Ncols) continue;
        float bv = bias ? bias[n] : 0.f;
#pragma unroll
        for (int mi = 0; mi < 2; ++mi) {
#pragma unroll
            for (int v = 0; v < 8; ++v) {
                long m = mBase + mi * 16 + v + 8 * lh;
                if (m >= M) continue;
                float x = acc[mi][t][v] + bv;
                if (relu) x = fmaxf(x, 0.f);
                if (out16) out16[m * (long)ldc + n] = (_Float16)x;
                else       out32[m * (long)ldc + n] = x;
            }
        }
    }
}

extern "C" void kernel_launch(void* const* d_in, const int* in_sizes, int n_in,
                              void* d_out, int out_size, void* d_ws, size_t ws_size,
                              hipStream_t stream) {
    const int*   so     = (const int*)d_in[1];
    const float* ggnn   = (const float*)d_in[2];
    const float* matrix = (const float*)d_in[3];
    const float* w3w = (const float*)d_in[4];
    const float* b3w = (const float*)d_in[5];
    const float* w3u = (const float*)d_in[6];
    const float* b3u = (const float*)d_in[7];
    const float* w4w = (const float*)d_in[8];
    const float* b4w = (const float*)d_in[9];
    const float* w5w = (const float*)d_in[10];
    const float* b5w = (const float*)d_in[11];
    const float* w5u = (const float*)d_in[12];
    const float* b5u = (const float*)d_in[13];
    const float* wout = (const float*)d_in[14];
    const float* bout = (const float*)d_in[15];
    const float* wcls = (const float*)d_in[16];
    const float* bcls = (const float*)d_in[17];
    float* out = (float*)d_out;

    char* ws = (char*)d_ws;
    size_t off = 0;
    auto alloc = [&](size_t bytes) -> void* {
        void* p = ws + off;
        off += (bytes + 255) & ~(size_t)255;
        return p;
    };
    const size_t actB = (size_t)RN_ * D_ * sizeof(_Float16);
    _Float16* h16   = (_Float16*)alloc(actB);
    _Float16* in16  = (_Float16*)alloc(actB);
    _Float16* av16  = (_Float16*)alloc(actB);
    _Float16* zv16  = (_Float16*)alloc(actB);
    _Float16* rh16  = (_Float16*)alloc(actB);
    _Float16* W3s   = (_Float16*)alloc((size_t)D_ * D_ * 2);
    _Float16* W4s   = (_Float16*)alloc((size_t)D_ * D_ * 2);
    _Float16* W5s   = (_Float16*)alloc((size_t)D_ * D_ * 2);
    _Float16* W3u16 = (_Float16*)alloc((size_t)D_ * D_ * 2);
    _Float16* W5u16 = (_Float16*)alloc((size_t)D_ * D_ * 2);
    _Float16* Wo16  = (_Float16*)alloc((size_t)D_ * 2 * D_ * 2);
    _Float16* Wc16  = (_Float16*)alloc((size_t)C_ * ND_ * 2);
    float*    Bg    = (float*)alloc((size_t)R_ * C_ * 4);

    // ---- prep ----
    k_gather<<<(R_ * C_ + 255) / 256, 256, 0, stream>>>(so, matrix, Bg);
    const int fb = (D_ * D_ + 255) / 256;
    k_fold<<<fb, 256, 0, stream>>>(w3w, W3s);
    k_fold<<<fb, 256, 0, stream>>>(w4w, W4s);
    k_fold<<<fb, 256, 0, stream>>>(w5w, W5s);
    auto cvt = [&](const float* s, _Float16* d, long n) {
        k_cvt<<<(int)((n + 255) / 256), 256, 0, stream>>>(s, d, n);
    };
    cvt(w3u, W3u16, (long)D_ * D_);
    cvt(w5u, W5u16, (long)D_ * D_);
    cvt(wout, Wo16, (long)D_ * 2 * D_);
    cvt(wcls, Wc16, (long)C_ * ND_);
    cvt(ggnn, h16, (long)RN_ * D_);
    cvt(ggnn, in16, (long)RN_ * D_);

    // ---- 3 GRU message-passing steps ----
    dim3 blk(256);
    dim3 grdMain(RN_ / 128, D_ / 64);   // (848, 8): 8 waves/block, 32x32 tile/wave
    for (int t = 0; t < 3; ++t) {
        k_av<<<dim3(R_, D_ / 256), blk, 0, stream>>>(h16, Bg, av16);
        k_gates<<<grdMain, blk, 0, stream>>>(av16, h16, W3s, W4s, W3u16,
                                             b3w, b3u, b4w, zv16, rh16);
        k_update<<<grdMain, blk, 0, stream>>>(av16, rh16, W5s, W5u16,
                                              b5w, b5u, zv16, h16);
    }

    // ---- final: relu([h, input] @ wout.T + bout), virtual concat via split-A ----
    k_gemm<<<grdMain, blk, 0, stream>>>(h16, in16, D_, D_, Wo16, 2 * D_, bout,
                                        rh16, nullptr, D_, RN_, D_, 2 * D_, 1);

    // ---- classifier: (R, 27136) @ wcls.T + bcls -> (R, 51) f32 ----
    k_gemm<<<dim3(R_ / 128, 1), blk, 0, stream>>>(rh16, rh16, ND_, ND_, Wc16, ND_, bcls,
                                                  nullptr, out, C_, R_, C_, ND_, 0);
}
